// TCR_79087527789205
// MI455X (gfx1250) — matrix operand order repaired and checked by reading, not compiled, not run on hardware
//
#include <hip/hip_runtime.h>
#include <hip/hip_bf16.h>
#include <math.h>

// ---------------------------------------------------------------------------
// Types for WMMA (gfx1250, wave32)
// ---------------------------------------------------------------------------
typedef __attribute__((ext_vector_type(16))) __bf16 v16bf;
typedef __attribute__((ext_vector_type(8)))  float  v8f;
typedef __attribute__((ext_vector_type(4)))  float  f4v;
typedef __attribute__((ext_vector_type(4)))  int    i4v;

union Frag { i4v q[2]; v16bf v; };           // 32B: one 16-bit A/B fragment

#define TDIM 2048
#define DDIM 1024
#define ALPHA 0.05f
#define EPSF  1e-16f
#define BVAL  (1.0f / 2048.0f)   // target marginal b
#define AVAL  (1.0f / 2048.0f)   // source marginal a

// ---------------------------------------------------------------------------
// CDNA5 async global->LDS copy (ASYNCcnt path), 32 bytes per call.
// copies gaddr + off into LDS at lds_addr + off; INST_OFFSET applies to both sides.
// ---------------------------------------------------------------------------
__device__ __forceinline__ void async_load_32B(unsigned int lds_addr,
                                               const __bf16* gaddr) {
    asm volatile("global_load_async_to_lds_b128 %0, %1, off"
                 :: "v"(lds_addr), "v"(gaddr) : "memory");
    asm volatile("global_load_async_to_lds_b128 %0, %1, off offset:16"
                 :: "v"(lds_addr), "v"(gaddr) : "memory");
}
__device__ __forceinline__ void wait_async0() {
    asm volatile("s_wait_asynccnt 0x0" ::: "memory");
}

// ---------------------------------------------------------------------------
// fp32 -> bf16 streaming convert (one quad per thread, 8B packed stores)
// ---------------------------------------------------------------------------
__global__ __launch_bounds__(256)
void cvt_bf16_kernel(const float* __restrict__ in, __bf16* __restrict__ out) {
    const int i = blockIdx.x * blockDim.x + threadIdx.x;   // quad index
    const f4v x = reinterpret_cast<const f4v*>(in)[i];
    union { __bf16 h[4]; unsigned long long q; } p;
    p.h[0] = (__bf16)x.x; p.h[1] = (__bf16)x.y;
    p.h[2] = (__bf16)x.z; p.h[3] = (__bf16)x.w;
    reinterpret_cast<unsigned long long*>(out)[i] = p.q;
}

// ---------------------------------------------------------------------------
// Row squared-norms (full fp32 from original data): one wave per row.
// ---------------------------------------------------------------------------
__global__ __launch_bounds__(256)
void rownorm_kernel(const float* __restrict__ A, float* __restrict__ out) {
    const int lane = threadIdx.x & 31;
    const int wave = threadIdx.x >> 5;
    const int row  = blockIdx.x * 8 + wave;
    const f4v* p = reinterpret_cast<const f4v*>(A + (size_t)row * DDIM);
    float s = 0.0f;
    for (int j = lane; j < DDIM / 4; j += 32) {
        f4v x = p[j];
        s += x.x * x.x + x.y * x.y + x.z * x.z + x.w * x.w;
    }
    for (int off = 16; off > 0; off >>= 1) s += __shfl_down(s, off);
    if (lane == 0) out[row] = s;
}

// ---------------------------------------------------------------------------
// WMMA GEMM: G = Xb @ Yb^T (bf16, f32 acc), double-buffered async-LDS tiles.
// Block tile 128x128, BK=32, 8 waves (4x2), each wave = 32x64 (2x4 WMMA tiles).
// Epilogue: d2 = |x|^2 + |y|^2 - 2*dot ; M = sqrt(max(d2,1e-12)) ; K = exp(-a*M)
// ---------------------------------------------------------------------------
__global__ __launch_bounds__(256)
void gemm_dist_kernel(const __bf16* __restrict__ Xb, const __bf16* __restrict__ Yb,
                      const float* __restrict__ xn, const float* __restrict__ yn,
                      float* __restrict__ Mout, float* __restrict__ Kout) {
    __shared__ __bf16 As[2][128 * 32];   // [buf][row][k], 8 KB each
    __shared__ __bf16 Bs[2][128 * 32];   // [buf][col][k], 8 KB each

    const int tid  = threadIdx.x;
    const int lane = tid & 31;
    const int wave = tid >> 5;
    const int wm   = wave & 3;           // row offset wm*32
    const int wn   = wave >> 2;          // col offset wn*64
    const int ln   = lane & 15;
    const int rowBase = blockIdx.x * 128;
    const int colBase = blockIdx.y * 128;

    // 16-bit fragment K-striping (ISA 7.12.2):
    // A (16x32): lanes 0-15 hold K {0..7,16..23}, lanes 16-31 hold K {8..15,24..31}
// B (32x16): lanes 0-15 hold K {0..7,16..23}, lanes 16-31 hold K {8..15,24..31}
    const int kbA = (lane < 16) ? 0 : 8;
    const int kbB = (lane < 16) ? 0 : 8;

    // tile loader mapping: each thread owns one 32B (16 bf16) chunk per slab
    const int ldr = tid >> 1;            // 0..127
    const int ldc = (tid & 1) * 16;      // 0 or 16

    const __bf16* gA = Xb + (size_t)(rowBase + ldr) * DDIM + ldc;
    const __bf16* gB = Yb + (size_t)(colBase + ldr) * DDIM + ldc;
    const unsigned int la0 = (unsigned int)(uintptr_t)&As[0][ldr * 32 + ldc];
    const unsigned int la1 = (unsigned int)(uintptr_t)&As[1][ldr * 32 + ldc];
    const unsigned int lb0 = (unsigned int)(uintptr_t)&Bs[0][ldr * 32 + ldc];
    const unsigned int lb1 = (unsigned int)(uintptr_t)&Bs[1][ldr * 32 + ldc];

    // prologue: stage k=0 into buffer 0
    async_load_32B(la0, gA);
    async_load_32B(lb0, gB);

    v8f acc[2][4] = {};

    for (int k0 = 0; k0 < DDIM; k0 += 32) {
        const int cur = (k0 >> 5) & 1;
        wait_async0();        // my async writes for buffer `cur` have landed
        __syncthreads();      // everyone's have; also all reads of `cur^1` retired

        if (k0 + 32 < DDIM) { // prefetch next slab into the other buffer
            async_load_32B(cur ? la0 : la1, gA + k0 + 32);
            async_load_32B(cur ? lb0 : lb1, gB + k0 + 32);
        }

        const __bf16* as = As[cur];
        const __bf16* bs = Bs[cur];
        Frag fa[2], fb[4];
#pragma unroll
        for (int mt = 0; mt < 2; ++mt) {
            const int r = wm * 32 + mt * 16 + ln;
            const i4v* p = reinterpret_cast<const i4v*>(&as[r * 32 + kbA]);
            fa[mt].q[0] = p[0];   // K kbA..kbA+7
            fa[mt].q[1] = p[2];   // K kbA+16..kbA+23
        }
#pragma unroll
        for (int nt = 0; nt < 4; ++nt) {
            const int c = wn * 64 + nt * 16 + ln;
            const i4v* p = reinterpret_cast<const i4v*>(&bs[c * 32 + kbB]);
            fb[nt].q[0] = p[0];   // K kbB..kbB+7
            fb[nt].q[1] = p[2];   // K kbB+16..kbB+23
        }

#pragma unroll
        for (int mt = 0; mt < 2; ++mt)
#pragma unroll
            for (int nt = 0; nt < 4; ++nt)
                acc[mt][nt] = __builtin_amdgcn_wmma_f32_16x16x32_bf16(
                    false, fa[mt].v, false, fb[nt].v,
                    (short)0, acc[mt][nt], false, false);
    }

    // ---- epilogue: distance + Gibbs kernel ----
    const int laneHalf = (lane < 16) ? 0 : 8;
#pragma unroll
    for (int mt = 0; mt < 2; ++mt)
#pragma unroll
        for (int nt = 0; nt < 4; ++nt) {
            const v8f a = acc[mt][nt];
            const int col = colBase + wn * 64 + nt * 16 + ln;
            const float ync = yn[col];
#pragma unroll
            for (int r = 0; r < 8; ++r) {
                const int row = rowBase + wm * 32 + mt * 16 + r + laneHalf;
                const float d2 = xn[row] + ync - 2.0f * a[r];
                const float m  = sqrtf(fmaxf(d2, 1e-12f));
                const size_t idx = (size_t)row * TDIM + col;
                Mout[idx] = m;
                Kout[idx] = __expf(-ALPHA * m);
            }
        }
}

// ---------------------------------------------------------------------------
// Sinkhorn state init: u = 1/T, v = 1, done = 0
// ---------------------------------------------------------------------------
__global__ void init_kernel(float* __restrict__ u, float* __restrict__ v,
                            int* __restrict__ done) {
    const int i = blockIdx.x * blockDim.x + threadIdx.x;
    if (i < TDIM) { u[i] = AVAL; v[i] = 1.0f; }
    if (i == 0) *done = 0;
}

// ---------------------------------------------------------------------------
// v = b / (K^T u + eps).  Block: 16 columns x 16 row-parallel, grid 128.
// ---------------------------------------------------------------------------
__global__ __launch_bounds__(256)
void sinkhorn_v_kernel(const float* __restrict__ Kmat, const float* __restrict__ u,
                       float* __restrict__ v, const int* __restrict__ done) {
    if (*done) return;
    __shared__ float red[256];
    const int tid = threadIdx.x;
    const int cl  = tid & 15;
    const int rp  = tid >> 4;
    const int c   = blockIdx.x * 16 + cl;
    float s = 0.0f;
    for (int t = rp; t < TDIM; t += 16)
        s += Kmat[t * TDIM + c] * u[t];
    red[tid] = s;
    __syncthreads();
    for (int off = 128; off >= 16; off >>= 1) {
        if (tid < off) red[tid] += red[tid + off];
        __syncthreads();
    }
    if (tid < 16) v[c] = BVAL / (red[tid] + EPSF);
}

// ---------------------------------------------------------------------------
// u = a / (K v + eps).  Block: 16 rows x 16 col-parallel, grid 128.
// ---------------------------------------------------------------------------
__global__ __launch_bounds__(256)
void sinkhorn_u_kernel(const float* __restrict__ Kmat, const float* __restrict__ v,
                       float* __restrict__ u, const int* __restrict__ done) {
    if (*done) return;
    __shared__ float red[256];
    const int tid = threadIdx.x;
    const int cp  = tid & 15;
    const int rl  = tid >> 4;
    const int r   = blockIdx.x * 16 + rl;
    float s = 0.0f;
    for (int c = cp; c < TDIM; c += 16)
        s += Kmat[r * TDIM + c] * v[c];
    red[tid] = s;
    __syncthreads();
    for (int off = 8; off >= 1; off >>= 1) {
        if ((tid & 15) < off) red[tid] += red[tid + off];
        __syncthreads();
    }
    if ((tid & 15) == 0) u[r] = AVAL / (red[tid] + EPSF);
}

// ---------------------------------------------------------------------------
// Error check (i % 50 == 0): bb = v .* (K^T u); err partial = sum |bb - b|
// ---------------------------------------------------------------------------
__global__ __launch_bounds__(256)
void err_partial_kernel(const float* __restrict__ Kmat, const float* __restrict__ u,
                        const float* __restrict__ v, const int* __restrict__ done,
                        float* __restrict__ errp) {
    if (*done) return;   // errp keeps values from the converged check (err <= thr)
    __shared__ float red[256];
    const int tid = threadIdx.x;
    const int cl  = tid & 15;
    const int rp  = tid >> 4;
    const int c   = blockIdx.x * 16 + cl;
    float s = 0.0f;
    for (int t = rp; t < TDIM; t += 16)
        s += Kmat[t * TDIM + c] * u[t];
    red[tid] = s;
    __syncthreads();
    for (int off = 128; off >= 16; off >>= 1) {
        if (tid < off) red[tid] += red[tid + off];
        __syncthreads();
    }
    if (tid == 0) {
        float e = 0.0f;
        for (int j = 0; j < 16; ++j)
            e += fabsf(v[blockIdx.x * 16 + j] * red[j] - BVAL);
        errp[blockIdx.x] = e;
    }
}

__global__ void err_reduce_kernel(const float* __restrict__ errp,
                                  int* __restrict__ done) {
    __shared__ float red[128];
    const int tid = threadIdx.x;
    red[tid] = errp[tid];
    __syncthreads();
    for (int off = 64; off >= 1; off >>= 1) {
        if (tid < off) red[tid] += red[tid + off];
        __syncthreads();
    }
    if (tid == 0 && red[0] <= 0.005f) *done = 1;   // sticky
}

// ---------------------------------------------------------------------------
// loss = sum u[t] * K[t,c] * v[c] * M[t,c]  (two-phase deterministic reduce)
// ---------------------------------------------------------------------------
__global__ __launch_bounds__(256)
void loss_partial_kernel(const float* __restrict__ Mmat, const float* __restrict__ Kmat,
                         const float* __restrict__ u, const float* __restrict__ v,
                         float* __restrict__ lossp) {
    __shared__ float red[256];
    const int tid = threadIdx.x;
    const int r0  = blockIdx.x * 8;
    float s = 0.0f;
    for (int rr = 0; rr < 8; ++rr) {
        const int r = r0 + rr;
        const float ur = u[r];
        const size_t base = (size_t)r * TDIM;
        for (int c = tid; c < TDIM; c += 256)
            s += ur * Kmat[base + c] * v[c] * Mmat[base + c];
    }
    red[tid] = s;
    __syncthreads();
    for (int off = 128; off >= 1; off >>= 1) {
        if (tid < off) red[tid] += red[tid + off];
        __syncthreads();
    }
    if (tid == 0) lossp[blockIdx.x] = red[0];
}

__global__ void loss_final_kernel(const float* __restrict__ lossp,
                                  float* __restrict__ out) {
    __shared__ float red[256];
    const int tid = threadIdx.x;
    red[tid] = lossp[tid];
    __syncthreads();
    for (int off = 128; off >= 1; off >>= 1) {
        if (tid < off) red[tid] += red[tid + off];
        __syncthreads();
    }
    if (tid == 0) out[0] = red[0];   // WEIGHT_LOSS_TCR == 1.0
}

// ---------------------------------------------------------------------------
// Launch
// ---------------------------------------------------------------------------
extern "C" void kernel_launch(void* const* d_in, const int* in_sizes, int n_in,
                              void* d_out, int out_size, void* d_ws, size_t ws_size,
                              hipStream_t stream) {
    (void)in_sizes; (void)n_in; (void)out_size; (void)ws_size;
    const float* X = (const float*)d_in[0];   // topic_emb      [2048,1024]
    const float* Y = (const float*)d_in[1];   // cluster_center [2048,1024]

    float*  W     = (float*)d_ws;
    float*  Mmat  = W;                          // 4,194,304 floats (16 MB)
    float*  Kmat  = W + 4194304;                // 4,194,304 floats (16 MB)
    __bf16* Xb    = (__bf16*)(W + 8388608);     // 2,097,152 bf16 (4 MB)
    __bf16* Yb    = (__bf16*)(W + 9437184);     // 2,097,152 bf16 (4 MB)
    float*  xn    = W + 10485760;               // 2048
    float*  yn    = xn + TDIM;                  // 2048
    float*  u     = yn + TDIM;                  // 2048
    float*  v     = u + TDIM;                   // 2048
    float*  errp  = v + TDIM;                   // 128
    float*  lossp = errp + 128;                 // 256
    int*    done  = (int*)(lossp + 256);        // 1

    cvt_bf16_kernel<<<2048, 256, 0, stream>>>(X, Xb);
    cvt_bf16_kernel<<<2048, 256, 0, stream>>>(Y, Yb);
    rownorm_kernel<<<256, 256, 0, stream>>>(X, xn);
    rownorm_kernel<<<256, 256, 0, stream>>>(Y, yn);
    gemm_dist_kernel<<<dim3(16, 16), 256, 0, stream>>>(Xb, Yb, xn, yn, Mmat, Kmat);
    init_kernel<<<8, 256, 0, stream>>>(u, v, done);

    for (int i = 0; i < 1000; ++i) {
        sinkhorn_v_kernel<<<128, 256, 0, stream>>>(Kmat, u, v, done);
        sinkhorn_u_kernel<<<128, 256, 0, stream>>>(Kmat, v, u, done);
        if (i % 50 == 0) {   // reference: ((i+1) % 50) == 1
            err_partial_kernel<<<128, 256, 0, stream>>>(Kmat, u, v, done, errp);
            err_reduce_kernel<<<1, 128, 0, stream>>>(errp, done);
        }
    }

    loss_partial_kernel<<<256, 256, 0, stream>>>(Mmat, Kmat, u, v, lossp);
    loss_final_kernel<<<1, 256, 0, stream>>>(lossp, (float*)d_out);
}
